// My_text_cla_81535659147879
// MI455X (gfx1250) — compile-verified, matrix operand order checked by reading, not run on hardware
//
#include <hip/hip_runtime.h>
#include <hip/hip_bf16.h>

typedef __attribute__((ext_vector_type(16))) __bf16 v16bf;
typedef __attribute__((ext_vector_type(8)))  float  v8f;

#define TM 128
#define TN 128
#define TK 32
#define LSTR 48   // bf16 elems per LDS row: 32 data + 16 pad => 96B rows, 32B-aligned k-halves

__device__ __forceinline__ __bf16 tobf(float f) { return (__bf16)f; }

// C[M,N] = act( alpha * A[M,K] x B + bias[N] )
// TRANSB=true : B stored [N,K] row-major (ldb = row stride)   -> A*B^T
// TRANSB=false: B stored [K,N] row-major                      -> A*B
// Staging uses clamped indices (no exec-divergent loads); OOB rows/cols only
// ever feed outputs that are masked at the epilogue store, so this is safe.
template<bool TRANSB>
__global__ __launch_bounds__(256)
void gemm_wmma(const float* __restrict__ A, const float* __restrict__ Bm,
               const float* __restrict__ bias, float* __restrict__ C,
               int M, int N, int K, int lda, int ldb, int ldc,
               long long sAo, long long sBo, long long sCo,
               float alpha, int doRelu)
{
    __shared__ __bf16 lA[TM * LSTR];
    __shared__ __bf16 lB[TN * LSTR];

    const long long bz = blockIdx.z;
    A  += bz * sAo;
    Bm += bz * sBo;
    C  += bz * sCo;

    const int m0   = blockIdx.y * TM;
    const int n0   = blockIdx.x * TN;
    const int tid  = threadIdx.x;
    const int lane = tid & 31;
    const int wid  = tid >> 5;
    const int wm   = (wid & 3) * 32;   // wave's M offset in tile
    const int wn   = (wid >> 2) * 64;  // wave's N offset in tile
    const int lrow = lane & 15;
    const int khalf = (lane >> 4) << 4;  // 0 or 16

    // staging index helpers (per thread)
    const int cb = (tid & 7) * 4;    // k-offset within tile, A / B-NT path
    const int rb = tid >> 3;         // row 0..31 (+{0,32,64,96})
    const int nb = (tid & 31) * 4;   // n-offset, B-NN path
    const int kb = tid >> 5;         // k 0..7 (+{0,8,16,24}), B-NN path

    v8f acc[2][4] = {};
    float4 ra[4], rbv[4];

    auto loadA = [&](int k0) {
        #pragma unroll
        for (int r = 0; r < 4; ++r) {
            int gm = m0 + rb + r * 32;
            if (gm > M - 1) gm = M - 1;                 // clamp, not branch
            ra[r] = *(const float4*)(A + (long long)gm * lda + (k0 + cb));
        }
    };
    auto loadB = [&](int k0) {
        if (TRANSB) {
            #pragma unroll
            for (int r = 0; r < 4; ++r) {
                int gn = n0 + rb + r * 32;
                if (gn > N - 1) gn = N - 1;
                rbv[r] = *(const float4*)(Bm + (long long)gn * ldb + (k0 + cb));
            }
        } else {
            int gn = n0 + nb;
            if (gn > N - 4) gn = N - 4;                 // N is always a multiple of 4
            #pragma unroll
            for (int r = 0; r < 4; ++r) {
                const int k = kb + r * 8;
                rbv[r] = *(const float4*)(Bm + (long long)(k0 + k) * ldb + gn);
            }
        }
    };
    auto stage = [&]() {
        #pragma unroll
        for (int r = 0; r < 4; ++r) {
            __bf16* d = &lA[(rb + r * 32) * LSTR + cb];
            d[0] = tobf(ra[r].x); d[1] = tobf(ra[r].y);
            d[2] = tobf(ra[r].z); d[3] = tobf(ra[r].w);
        }
        if (TRANSB) {
            #pragma unroll
            for (int r = 0; r < 4; ++r) {
                __bf16* d = &lB[(rb + r * 32) * LSTR + cb];
                d[0] = tobf(rbv[r].x); d[1] = tobf(rbv[r].y);
                d[2] = tobf(rbv[r].z); d[3] = tobf(rbv[r].w);
            }
        } else {
            #pragma unroll
            for (int r = 0; r < 4; ++r) {
                const int k = kb + r * 8;
                lB[(nb + 0) * LSTR + k] = tobf(rbv[r].x);
                lB[(nb + 1) * LSTR + k] = tobf(rbv[r].y);
                lB[(nb + 2) * LSTR + k] = tobf(rbv[r].z);
                lB[(nb + 3) * LSTR + k] = tobf(rbv[r].w);
            }
        }
    };

    // software pipeline: global loads of tile k+1 overlap WMMA of tile k
    loadA(0); loadB(0);
    for (int k0 = 0; k0 < K; k0 += TK) {
        stage();
        __syncthreads();
        if (k0 + TK < K) { loadA(k0 + TK); loadB(k0 + TK); }

        // WMMA fragments: lanes 0-15 -> K 0..15, lanes 16-31 -> K 16..31
        v16bf aF[2], bF[4];
        #pragma unroll
        for (int i = 0; i < 2; ++i)
            aF[i] = *(const v16bf*)&lA[(wm + i * 16 + lrow) * LSTR + khalf];
        #pragma unroll
        for (int j = 0; j < 4; ++j)
            bF[j] = *(const v16bf*)&lB[(wn + j * 16 + lrow) * LSTR + khalf];

        #pragma unroll
        for (int i = 0; i < 2; ++i)
            #pragma unroll
            for (int j = 0; j < 4; ++j)
                acc[i][j] = __builtin_amdgcn_wmma_f32_16x16x32_bf16(
                    false, aF[i], false, bF[j], (short)0, acc[i][j], false, false);
        __syncthreads();
    }

    // epilogue: C/D layout: VGPR r, lanes 0-15 -> M=r, lanes 16-31 -> M=r+8
    #pragma unroll
    for (int i = 0; i < 2; ++i) {
        #pragma unroll
        for (int j = 0; j < 4; ++j) {
            const int n = n0 + wn + j * 16 + lrow;
            if (n >= N) continue;
            const int mB = m0 + wm + i * 16 + ((lane < 16) ? 0 : 8);
            const float bv = bias ? bias[n] : 0.f;
            #pragma unroll
            for (int r = 0; r < 8; ++r) {
                const int m = mB + r;
                if (m < M) {
                    float val = acc[i][j][r] * alpha + bv;
                    if (doRelu) val = fmaxf(val, 0.f);
                    C[(long long)m * ldc + n] = val;
                }
            }
        }
    }
}

// row-wise softmax, one block (256 thr) per row, cols multiple of 256 (<=2048)
__global__ __launch_bounds__(256)
void softmax_rows(float* __restrict__ X, int cols)
{
    float* row = X + (long long)blockIdx.x * cols;
    __shared__ float red[256];
    const int tid = threadIdx.x;
    const int per = cols >> 8;
    float v[8];
    float mx = -3.4e38f;
    for (int i = 0; i < per; ++i) { v[i] = row[tid + (i << 8)]; mx = fmaxf(mx, v[i]); }
    red[tid] = mx; __syncthreads();
    for (int s = 128; s > 0; s >>= 1) { if (tid < s) red[tid] = fmaxf(red[tid], red[tid + s]); __syncthreads(); }
    mx = red[0]; __syncthreads();
    float sum = 0.f;
    for (int i = 0; i < per; ++i) { v[i] = __expf(v[i] - mx); sum += v[i]; }
    red[tid] = sum; __syncthreads();
    for (int s = 128; s > 0; s >>= 1) { if (tid < s) red[tid] += red[tid + s]; __syncthreads(); }
    const float inv = 1.f / red[0]; __syncthreads();
    for (int i = 0; i < per; ++i) row[tid + (i << 8)] = v[i] * inv;
}

// out[r, 0..D) = LN(Xa[r] + Xb[r]) * g + b  ; one block per row
__global__ __launch_bounds__(256)
void add_ln(const float* __restrict__ Xa, const float* __restrict__ Xb,
            const float* __restrict__ g, const float* __restrict__ bb,
            float* __restrict__ out, int D, int ldOut)
{
    const long long r = blockIdx.x;
    const float* a = Xa + r * D;
    const float* b2 = Xb + r * D;
    float* o = out + r * ldOut;
    __shared__ float red[256];
    const int tid = threadIdx.x;
    const int per = D >> 8;
    float v[4];
    float s = 0.f;
    for (int i = 0; i < per; ++i) { const int c = tid + (i << 8); v[i] = a[c] + b2[c]; s += v[i]; }
    red[tid] = s; __syncthreads();
    for (int st = 128; st > 0; st >>= 1) { if (tid < st) red[tid] += red[tid + st]; __syncthreads(); }
    const float mean = red[0] / (float)D; __syncthreads();
    s = 0.f;
    for (int i = 0; i < per; ++i) { const float d = v[i] - mean; s += d * d; }
    red[tid] = s; __syncthreads();
    for (int st = 128; st > 0; st >>= 1) { if (tid < st) red[tid] += red[tid + st]; __syncthreads(); }
    const float inv = rsqrtf(red[0] / (float)D + 1e-5f); __syncthreads();
    for (int i = 0; i < per; ++i) {
        const int c = tid + (i << 8);
        o[c] = (v[i] - mean) * inv * g[c] + bb[c];
    }
}

// logits[b,c] = p_cla[b, S-1, :] . W_clf[c, :] + b_clf[c] ; block per batch, 4 waves
__global__ __launch_bounds__(128)
void classify_last(const float* __restrict__ pcla, const float* __restrict__ Wc,
                   const float* __restrict__ bc, float* __restrict__ logits,
                   int D, int S, int NC)
{
    const int b = blockIdx.x;
    const float* xr = pcla + ((long long)b * S + (S - 1)) * D;
    const int c = threadIdx.x >> 5;
    const int lane = threadIdx.x & 31;
    if (c >= NC) return;
    float s = 0.f;
    for (int d = lane; d < D; d += 32) s += xr[d] * Wc[(long long)c * D + d];
    for (int off = 16; off > 0; off >>= 1) s += __shfl_down(s, off, 32);
    if (lane == 0) logits[b * NC + c] = s + bc[c];
}

extern "C" void kernel_launch(void* const* d_in, const int* in_sizes, int n_in,
                              void* d_out, int out_size, void* d_ws, size_t ws_size,
                              hipStream_t stream)
{
    (void)in_sizes; (void)n_in; (void)out_size; (void)ws_size;
    constexpr int B = 16, S = 1024, IN = 768, OUT = 768, BF = 768, Z = 512,
                  L = 512, H = 8, FF = 1024, NC = 4, HD = 64;
    constexpr int DP = OUT + L;  // 1280 p_cla width

    const float* x       = (const float*)d_in[0];
    const float* z_dec   = (const float*)d_in[1];
    const float* clauses = (const float*)d_in[2];
    const float* W_cla   = (const float*)d_in[3];
    const float* b_cla   = (const float*)d_in[4];
    const float* W_att   = (const float*)d_in[5];
    const float* b_att   = (const float*)d_in[6];
    const float* W_z     = (const float*)d_in[7];
    const float* b_z     = (const float*)d_in[8];
    const float* in_w    = (const float*)d_in[9];
    const float* in_b    = (const float*)d_in[10];
    const float* out_w   = (const float*)d_in[11];
    const float* out_b   = (const float*)d_in[12];
    const float* ln1g    = (const float*)d_in[13];
    const float* ln1b    = (const float*)d_in[14];
    const float* ln2g    = (const float*)d_in[15];
    const float* ln2b    = (const float*)d_in[16];
    const float* ff1w    = (const float*)d_in[17];
    const float* ff1b    = (const float*)d_in[18];
    const float* ff2w    = (const float*)d_in[19];
    const float* ff2b    = (const float*)d_in[20];
    const float* W_clf   = (const float*)d_in[21];
    const float* b_clf   = (const float*)d_in[22];

    float* logits = (float*)d_out;
    float* pcla   = logits + B * NC;  // [B, S, 1280]

    char* ws = (char*)d_ws;
    const size_t SZ768 = (size_t)B * S * 768 * 4;   // 48MB
    const size_t SZSC  = (size_t)B * S * S * 4;     // 64MB
    const size_t SZ512 = (size_t)B * S * 512 * 4;   // 32MB
    const size_t SZQKV = (size_t)B * S * 1536 * 4;  // 96MB
    float* out0 = (float*)(ws);                                         // relu(x W_cla^T)
    float* t768 = (float*)(ws + SZ768);                                 // att / z / attn-out (reused)
    float* sc   = (float*)(ws + 2 * SZ768);                             // [B,S,S] scores
    float* cbuf = (float*)(ws + 2 * SZ768 + SZSC);                      // c [B,S,512]
    float* qkv  = (float*)(ws + 2 * SZ768 + SZSC + SZ512);              // [B,S,1536]
    float* hbuf = (float*)(ws + 2 * SZ768 + SZSC + SZ512 + SZQKV);      // ln1 out
    float* ffb  = (float*)(ws + 2 * SZ768 + SZSC + 2 * SZ512 + SZQKV);  // [B,S,1024]
    float* o2   = (float*)(ws + 2 * SZ768 + SZSC + 2 * SZ512 + SZQKV + (size_t)B * S * 1024 * 4);

    dim3 blk(256);
    auto grid = [](int m, int n, int bat) {
        return dim3((unsigned)((n + TN - 1) / TN), (unsigned)((m + TM - 1) / TM), (unsigned)bat);
    };
    const long long sS768 = (long long)S * 768, sSS = (long long)S * S;
    const long long sS512 = (long long)S * 512, sS1536 = (long long)S * 1536;

    // 1) out0 = relu(x W_cla^T + b_cla)
    gemm_wmma<true><<<grid(B * S, OUT, 1), blk, 0, stream>>>(
        x, W_cla, b_cla, out0, B * S, OUT, IN, IN, IN, OUT, 0, 0, 0, 1.f, 1);
    // 2) t768 = out0 W_att^T + b_att
    gemm_wmma<true><<<grid(B * S, OUT, 1), blk, 0, stream>>>(
        out0, W_att, b_att, t768, B * S, OUT, OUT, OUT, OUT, OUT, 0, 0, 0, 1.f, 0);
    // 3) sc[b] = out_att[b] out0[b]^T
    gemm_wmma<true><<<grid(S, S, B), blk, 0, stream>>>(
        t768, out0, nullptr, sc, S, S, OUT, OUT, OUT, S, sS768, sS768, sSS, 1.f, 0);
    // 4) softmax over last dim
    softmax_rows<<<dim3(B * S), blk, 0, stream>>>(sc, S);
    // 5) ctx[b] = sc[b] out0[b]  -> p_cla[:, :, 0:768]
    gemm_wmma<false><<<grid(S, OUT, B), blk, 0, stream>>>(
        sc, out0, nullptr, pcla, S, OUT, S, S, OUT, DP, sSS, sS768, (long long)S * DP, 1.f, 0);
    // 6) z = z_dec W_z^T + b_z  (reuse t768)
    gemm_wmma<true><<<grid(B * S, BF, 1), blk, 0, stream>>>(
        z_dec, W_z, b_z, t768, B * S, BF, Z, Z, Z, BF, 0, 0, 0, 1.f, 0);
    // 7) c[b] = z[b] clauses[b]^T
    gemm_wmma<true><<<grid(S, L, B), blk, 0, stream>>>(
        t768, clauses, nullptr, cbuf, S, L, BF, BF, BF, L, sS768, (long long)L * BF, sS512, 1.f, 0);
    // 8) qkv = c in_proj_w^T + in_proj_b
    gemm_wmma<true><<<grid(B * S, 3 * L, 1), blk, 0, stream>>>(
        cbuf, in_w, in_b, qkv, B * S, 3 * L, L, L, L, 3 * L, 0, 0, 0, 1.f, 0);
    // 9) 8-head attention (score scratch reused per head); t768 reused for attn out
    const float scale = 0.125f;  // 1/sqrt(64)
    for (int h = 0; h < H; ++h) {
        gemm_wmma<true><<<grid(S, S, B), blk, 0, stream>>>(
            qkv + h * HD, qkv + L + h * HD, nullptr, sc,
            S, S, HD, 3 * L, 3 * L, S, sS1536, sS1536, sSS, scale, 0);
        softmax_rows<<<dim3(B * S), blk, 0, stream>>>(sc, S);
        gemm_wmma<false><<<grid(S, HD, B), blk, 0, stream>>>(
            sc, qkv + 2 * L + h * HD, nullptr, t768 + h * HD,
            S, HD, S, S, 3 * L, L, sSS, sS1536, sS512, 1.f, 0);
    }
    // 10) o2 = attn_out out_proj^T + b
    gemm_wmma<true><<<grid(B * S, L, 1), blk, 0, stream>>>(
        t768, out_w, out_b, o2, B * S, L, L, L, L, L, 0, 0, 0, 1.f, 0);
    // 11) h = LN1(c + o2)
    add_ln<<<dim3(B * S), blk, 0, stream>>>(cbuf, o2, ln1g, ln1b, hbuf, L, L);
    // 12) ff = relu(h ff1^T + b)
    gemm_wmma<true><<<grid(B * S, FF, 1), blk, 0, stream>>>(
        hbuf, ff1w, ff1b, ffb, B * S, FF, L, L, L, FF, 0, 0, 0, 1.f, 1);
    // 13) ff2 = ff ff2^T + b  (reuse t768 as [B*S,512])
    gemm_wmma<true><<<grid(B * S, L, 1), blk, 0, stream>>>(
        ffb, ff2w, ff2b, t768, B * S, L, FF, FF, FF, L, 0, 0, 0, 1.f, 0);
    // 14) trans = LN2(h + ff2) -> p_cla[:, :, 768:1280]
    add_ln<<<dim3(B * S), blk, 0, stream>>>(hbuf, t768, ln2g, ln2b, pcla + OUT, L, DP);
    // 15) logits from last token
    classify_last<<<dim3(B), dim3(128), 0, stream>>>(pcla, W_clf, b_clf, logits, DP, S, NC);
}
